// SelfAttention_68728066670626
// MI455X (gfx1250) — matrix operand order repaired and checked by reading, not compiled, not run on hardware
//
#include <hip/hip_runtime.h>
#include <stdint.h>

#define DI __device__ __forceinline__

typedef __bf16 bf16_t;
typedef __attribute__((ext_vector_type(16))) __bf16   v16bf;
typedef __attribute__((ext_vector_type(8)))  float    v8f;
typedef __attribute__((ext_vector_type(4)))  float    f32x4;
typedef __attribute__((ext_vector_type(4)))  uint32_t u32x4;

constexpr int NB = 2, LL = 2048, EE = 1024, HH = 16, DD = 64;
constexpr int MR = NB * LL;               // 4096 flattened rows

DI v8f wmma_bf16(v16bf a, v16bf b, v8f c) {
  // (neg_a, A, neg_b, B, c_mod, C, reuse_a, reuse_b)
  return __builtin_amdgcn_wmma_f32_16x16x32_bf16(false, a, false, b, (short)0, c,
                                                 false, false);
}

// ---- fragment loaders -------------------------------------------------------
// A 16x32 bf16 layout: lane l -> row m0+(l&15); K = kb..kb+7 (elems 0..7) and
// kb+16..kb+23 (elems 8..15), kb = k0 + (l>>4)*8.
DI v16bf load_afrag(const float* __restrict__ base, int stride, int m0, int k0, int lane) {
  int r  = m0 + (lane & 15);
  int kb = k0 + ((lane >> 4) << 3);
  const float* p = base + (size_t)r * stride + kb;
  f32x4 x0 = *(const f32x4*)(p);
  f32x4 x1 = *(const f32x4*)(p + 4);
  f32x4 x2 = *(const f32x4*)(p + 16);
  f32x4 x3 = *(const f32x4*)(p + 20);
  v16bf a;
#pragma unroll
  for (int i = 0; i < 4; ++i) {
    a[i] = (bf16_t)x0[i]; a[4+i] = (bf16_t)x1[i];
    a[8+i] = (bf16_t)x2[i]; a[12+i] = (bf16_t)x3[i];
  }
  return a;
}

DI v16bf load_afrag(const bf16_t* __restrict__ base, int stride, int m0, int k0, int lane) {
  int r  = m0 + (lane & 15);
  int kb = k0 + ((lane >> 4) << 3);
  const bf16_t* p = base + (size_t)r * stride + kb;
  union { v16bf v; u32x4 u[2]; } f;
  f.u[0] = *(const u32x4*)(p);
  f.u[1] = *(const u32x4*)(p + 16);
  return f.v;
}

// B 32x16 bf16 layout: lane l -> col n0+(l&15); K = kb..kb+7 (elems 0..7) and
// kb+16..kb+23 (elems 8..15), kb = k0 + (l>>4)*8.
DI v16bf load_bfrag(const float* __restrict__ base, int stride, int n0, int k0, int lane) {
  int r  = n0 + (lane & 15);
  int kb = k0 + ((lane >> 4) << 3);
  const float* p = base + (size_t)r * stride + kb;
  v16bf b;
#pragma unroll
  for (int j = 0; j < 4; ++j) {
    f32x4 x = *(const f32x4*)(p + 4 * j + ((j >> 1) << 3));
#pragma unroll
    for (int i = 0; i < 4; ++i) b[4*j + i] = (bf16_t)x[i];
  }
  return b;
}

DI v16bf load_bfrag(const bf16_t* __restrict__ base, int stride, int n0, int k0, int lane) {
  int r  = n0 + (lane & 15);
  int kb = k0 + ((lane >> 4) << 3);
  const bf16_t* p = base + (size_t)r * stride + kb;
  union { v16bf v; u32x4 u[2]; } f;
  f.u[0] = *(const u32x4*)(p);
  f.u[1] = *(const u32x4*)(p + 16);
  return f.v;
}

// ---- one-shot f32 -> bf16 conversion (removes all cvt from GEMM hot loops) --
__global__ __launch_bounds__(256) void cvt_f32_bf16_kernel(const float* __restrict__ src,
                                                           bf16_t* __restrict__ dst, int n) {
  int i = (blockIdx.x * 256 + threadIdx.x) * 8;
  if (i >= n) return;
  f32x4 a = *(const f32x4*)(src + i);
  f32x4 b = *(const f32x4*)(src + i + 4);
  union { u32x4 u; bf16_t e[8]; } pk;
#pragma unroll
  for (int j = 0; j < 4; ++j) { pk.e[j] = (bf16_t)a[j]; pk.e[4+j] = (bf16_t)b[j]; }
  *(u32x4*)(dst + i) = pk.u;
}

// ---- unified GEMM: C[M,E] = A[M,E] * W[E,E]^T (64x64 tile / wave) -----------
// TRANS==1 (V path): store bf16 transposed [N,H,D,L]; else bf16 [M,E].
// OUTF32: fp32 output with bias (final projection).
template <int TRANS, bool OUTF32, typename AT, typename BT>
__global__ __launch_bounds__(256) void gemm_kernel(const AT* __restrict__ A,
                                                   const BT* __restrict__ W,
                                                   const float* __restrict__ bias,
                                                   bf16_t* __restrict__ outb,
                                                   float* __restrict__ outf) {
  int lane = threadIdx.x & 31;
  int wid  = blockIdx.x * (blockDim.x >> 5) + (threadIdx.x >> 5);
  constexpr int TN = EE / 64;                      // 16 n-tiles
  int tm = wid / TN, tn = wid % TN;
  if (tm >= MR / 64) return;
  int m0 = tm * 64, n0 = tn * 64;

  v8f acc[4][4] = {};
  for (int k0 = 0; k0 < EE; k0 += 32) {
    v16bf a[4], b[4];
#pragma unroll
    for (int i = 0; i < 4; ++i) a[i] = load_afrag(A, EE, m0 + 16*i, k0, lane);
#pragma unroll
    for (int j = 0; j < 4; ++j) b[j] = load_bfrag(W, EE, n0 + 16*j, k0, lane);
#pragma unroll
    for (int i = 0; i < 4; ++i)
#pragma unroll
      for (int j = 0; j < 4; ++j) acc[i][j] = wmma_bf16(a[i], b[j], acc[i][j]);
  }

  int nc = lane & 15, hi = lane >> 4;
#pragma unroll
  for (int i = 0; i < 4; ++i)
#pragma unroll
    for (int j = 0; j < 4; ++j) {
      int n = n0 + 16*j + nc;
      if (OUTF32) {
        float bv = bias[n];
#pragma unroll
        for (int v = 0; v < 8; ++v) {
          int m = m0 + 16*i + 8*hi + v;
          outf[(size_t)m * EE + n] = acc[i][j][v] + bv;
        }
      } else if (TRANS == 0) {
#pragma unroll
        for (int v = 0; v < 8; ++v) {
          int m = m0 + 16*i + 8*hi + v;
          outb[(size_t)m * EE + n] = (bf16_t)acc[i][j][v];
        }
      } else {
        int h = n >> 6, d = n & 63;
        int mfirst = m0 + 16*i + 8*hi;             // 8 consecutive positions
        int nb = mfirst >> 11, pos = mfirst & (LL - 1);
        union { u32x4 u; bf16_t e[8]; } pk;
#pragma unroll
        for (int v = 0; v < 8; ++v) pk.e[v] = (bf16_t)acc[i][j][v];
        *(u32x4*)(outb + ((size_t)((nb * HH + h) * DD + d)) * LL + pos) = pk.u;
      }
    }
}

// ---- flash attention (causal), one wave per (batch, head, 16-query tile) ----
__global__ __launch_bounds__(256) void flash_attn_kernel(const bf16_t* __restrict__ Qb,
                                                         const bf16_t* __restrict__ Kb,
                                                         const bf16_t* __restrict__ Vt,
                                                         bf16_t* __restrict__ Ob) {
  __shared__ bf16_t plds[8][16 * 64];              // per-wave P staging (16KB)
  int lane  = threadIdx.x & 31;
  int wslot = threadIdx.x >> 5;
  int wid   = blockIdx.x * (blockDim.x >> 5) + wslot;
  if (wid >= NB * HH * (LL / 16)) return;
  int qt = wid & 127, h = (wid >> 7) & 15, nb = wid >> 11;
  int q0 = qt * 16;

  const bf16_t* Qh = Qb + (size_t)nb * LL * EE + h * 64;
  const bf16_t* Kh = Kb + (size_t)nb * LL * EE + h * 64;
  const bf16_t* Vh = Vt + (size_t)(nb * HH + h) * DD * LL;

  v16bf qa0 = load_afrag(Qh, EE, q0, 0, lane);
  v16bf qa1 = load_afrag(Qh, EE, q0, 32, lane);

  v8f oacc[4] = {};
  float rm[8], rl[8];
#pragma unroll
  for (int v = 0; v < 8; ++v) { rm[v] = -3e38f; rl[v] = 0.f; }

  int nc = lane & 15, hi = lane >> 4;
  const float cexp = 0.125f * 1.44269504f;         // (1/sqrt(64)) * log2(e)
  bf16_t* pl = &plds[wslot][0];
  int nch = (q0 + 16 + 63) >> 6;                   // 64-key chunks

  for (int c = 0; c < nch; ++c) {
    int ks = c * 64;
    // ---- S = Q K^T (bf16 WMMA, fp32 acc) ----
    v8f s[4];
#pragma unroll
    for (int t = 0; t < 4; ++t) {
      v16bf b0 = load_bfrag(Kh, EE, ks + 16*t, 0, lane);
      v16bf b1 = load_bfrag(Kh, EE, ks + 16*t, 32, lane);
      v8f z = {};
      s[t] = wmma_bf16(qa1, b1, wmma_bf16(qa0, b0, z));
    }
    // ---- causal mask + online softmax (raw-score domain, exp2-based) ----
    float p[4][8];
#pragma unroll
    for (int v = 0; v < 8; ++v) {
      int q = q0 + 8*hi + v;
      float cm = -3e38f;
#pragma unroll
      for (int t = 0; t < 4; ++t) {
        int col = ks + 16*t + nc;
        float val = (col <= q) ? s[t][v] : -3e38f;
        p[t][v] = val;
        cm = fmaxf(cm, val);
      }
#pragma unroll
      for (int off = 1; off < 16; off <<= 1) cm = fmaxf(cm, __shfl_xor(cm, off, 32));
      float nm    = fmaxf(rm[v], cm);
      float alpha = __builtin_amdgcn_exp2f((rm[v] - nm) * cexp);
      rm[v] = nm;
      float rs = 0.f;
#pragma unroll
      for (int t = 0; t < 4; ++t) {
        float e = __builtin_amdgcn_exp2f((p[t][v] - nm) * cexp);
        p[t][v] = e; rs += e;
      }
#pragma unroll
      for (int off = 1; off < 16; off <<= 1) rs += __shfl_xor(rs, off, 32);
      rl[v] = rl[v] * alpha + rs;
#pragma unroll
      for (int dt = 0; dt < 4; ++dt) oacc[dt][v] *= alpha;
    }
    // ---- P: C-layout -> A-layout via LDS ----
#pragma unroll
    for (int t = 0; t < 4; ++t)
#pragma unroll
      for (int v = 0; v < 8; ++v)
        pl[(8*hi + v) * 64 + 16*t + nc] = (bf16_t)p[t][v];
    asm volatile("s_wait_dscnt 0" ::: "memory");   // wave-private LDS RAW fence
    union { v16bf v; u32x4 u[2]; } pa0, pa1;
    {
      const bf16_t* pr = pl + (lane & 15) * 64 + ((lane >> 4) << 3);
      pa0.u[0] = *(const u32x4*)(pr);
      pa0.u[1] = *(const u32x4*)(pr + 16);
      pa1.u[0] = *(const u32x4*)(pr + 32);
      pa1.u[1] = *(const u32x4*)(pr + 48);
    }
    // ---- O += P V ----
#pragma unroll
    for (int dt = 0; dt < 4; ++dt) {
      v16bf vb0 = load_bfrag(Vh, LL, dt * 16, ks, lane);
      v16bf vb1 = load_bfrag(Vh, LL, dt * 16, ks + 32, lane);
      oacc[dt] = wmma_bf16(pa1.v, vb1, wmma_bf16(pa0.v, vb0, oacc[dt]));
    }
  }
  // ---- normalize + store O (bf16 [N,L,E]) ----
#pragma unroll
  for (int dt = 0; dt < 4; ++dt) {
    int n = h * 64 + dt * 16 + nc;
#pragma unroll
    for (int v = 0; v < 8; ++v) {
      int m = nb * LL + q0 + 8*hi + v;
      Ob[(size_t)m * EE + n] = (bf16_t)(oacc[dt][v] / rl[v]);
    }
  }
}

// ---- host-side launch -------------------------------------------------------
extern "C" void kernel_launch(void* const* d_in, const int* in_sizes, int n_in,
                              void* d_out, int out_size, void* d_ws, size_t ws_size,
                              hipStream_t stream) {
  const float* values = (const float*)d_in[0];
  const float* keys   = (const float*)d_in[1];
  const float* query  = (const float*)d_in[2];
  // d_in[3] = mask (causal, implied — unused)
  const float* Wv = (const float*)d_in[4];
  const float* Wk = (const float*)d_in[5];
  const float* Wq = (const float*)d_in[6];
  const float* Wo = (const float*)d_in[7];
  const float* bo = (const float*)d_in[8];
  float* out = (float*)d_out;

  const size_t actE = (size_t)MR * EE;                 // 4M elems
  const size_t wE   = (size_t)EE * EE;                 // 1M elems
  // core workspace: Q, K (bf16 [N,L,E]), Vt (bf16 [N,H,D,L]), O (bf16 [N,L,E])
  bf16_t* Qb = (bf16_t*)d_ws;
  bf16_t* Kb = Qb + actE;
  bf16_t* Vt = Kb + actE;
  bf16_t* Ob = Vt + actE;
  // preconverted bf16 copies of activations + weights (if workspace allows)
  bf16_t* vB = Ob + actE;
  bf16_t* kB = vB + actE;
  bf16_t* qB = kB + actE;
  bf16_t* WvB = qB + actE;
  bf16_t* WkB = WvB + wE;
  bf16_t* WqB = WkB + wE;
  bf16_t* WoB = WqB + wE;
  const size_t need_full = (7 * actE + 4 * wE) * sizeof(bf16_t);   // 64 MB

  const int proj_blocks = (MR / 64) * (EE / 64) / 8;   // 128
  const int attn_blocks = NB * HH * (LL / 16) / 8;     // 512
  const int act_cvt_blocks = (int)(actE / (256 * 8));  // 2048
  const int w_cvt_blocks   = (int)(wE / (256 * 8));    // 512

  if (ws_size >= need_full) {
    // one-shot conversions, then pure-bf16 GEMM hot loops (no cvt per k-step)
    cvt_f32_bf16_kernel<<<act_cvt_blocks, 256, 0, stream>>>(values, vB, (int)actE);
    cvt_f32_bf16_kernel<<<act_cvt_blocks, 256, 0, stream>>>(keys,   kB, (int)actE);
    cvt_f32_bf16_kernel<<<act_cvt_blocks, 256, 0, stream>>>(query,  qB, (int)actE);
    cvt_f32_bf16_kernel<<<w_cvt_blocks,   256, 0, stream>>>(Wv, WvB, (int)wE);
    cvt_f32_bf16_kernel<<<w_cvt_blocks,   256, 0, stream>>>(Wk, WkB, (int)wE);
    cvt_f32_bf16_kernel<<<w_cvt_blocks,   256, 0, stream>>>(Wq, WqB, (int)wE);
    cvt_f32_bf16_kernel<<<w_cvt_blocks,   256, 0, stream>>>(Wo, WoB, (int)wE);

    gemm_kernel<0, false, bf16_t, bf16_t><<<proj_blocks, 256, 0, stream>>>(qB, WqB, nullptr, Qb, nullptr);
    gemm_kernel<0, false, bf16_t, bf16_t><<<proj_blocks, 256, 0, stream>>>(kB, WkB, nullptr, Kb, nullptr);
    gemm_kernel<1, false, bf16_t, bf16_t><<<proj_blocks, 256, 0, stream>>>(vB, WvB, nullptr, Vt, nullptr);
    flash_attn_kernel<<<attn_blocks, 256, 0, stream>>>(Qb, Kb, Vt, Ob);
    gemm_kernel<0, true, bf16_t, bf16_t><<<proj_blocks, 256, 0, stream>>>(Ob, WoB, bo, nullptr, out);
  } else {
    // fallback: convert inside the GEMM loops (needs only 32 MB workspace)
    gemm_kernel<0, false, float, float><<<proj_blocks, 256, 0, stream>>>(query,  Wq, nullptr, Qb, nullptr);
    gemm_kernel<0, false, float, float><<<proj_blocks, 256, 0, stream>>>(keys,   Wk, nullptr, Kb, nullptr);
    gemm_kernel<1, false, float, float><<<proj_blocks, 256, 0, stream>>>(values, Wv, nullptr, Vt, nullptr);
    flash_attn_kernel<<<attn_blocks, 256, 0, stream>>>(Qb, Kb, Vt, Ob);
    gemm_kernel<0, true, bf16_t, float><<<proj_blocks, 256, 0, stream>>>(Ob, Wo, bo, nullptr, out);
  }
}
